// CATBlock_10479720202401
// MI455X (gfx1250) — compile-verified, matrix operand order checked by reading, not run on hardware
//
#include <hip/hip_runtime.h>
#include <cstdint>
#include <cstddef>

// ---------------------------------------------------------------------------
// CAT block (windowed attention, p=7, heads=12, d=32) for gfx1250 / MI455X.
// bf16 WMMA (V_WMMA_F32_16X16X32_BF16) for all GEMMs + attention.
// GEMM tiles staged with the Tensor Data Mover (TENSOR_LOAD_TO_LDS),
// double-buffered and synchronized with s_wait_tensorcnt.
// ---------------------------------------------------------------------------

typedef unsigned short u16;
typedef __attribute__((ext_vector_type(16))) __bf16 v16bf;
typedef __attribute__((ext_vector_type(8)))  float  v8f;
typedef __attribute__((ext_vector_type(4)))  unsigned int v4u;
typedef __attribute__((ext_vector_type(8)))  int v8i_t;
typedef __attribute__((ext_vector_type(4)))  int v4i_t;

struct __align__(16) U4 { unsigned int x, y, z, w; };
union Frag { v16bf v; U4 q[2]; u16 u[16]; };

// Problem constants (from setup_inputs)
#define MTOK 100352   // B * H * W = 32*56*56
#define CDIM 384
#define C3   1152
#define HID  1536
#define NWIN 2048     // 32 * 8 * 8
#define HEADS 12

#if defined(__has_builtin)
#if __has_builtin(__builtin_amdgcn_tensor_load_to_lds) && __has_builtin(__builtin_amdgcn_s_wait_tensorcnt)
#define USE_TDM 1
#endif
#endif
#ifndef USE_TDM
#define USE_TDM 0
#endif

__device__ __forceinline__ u16 f2bf(float f) {
    union { float f; unsigned u; } c; c.f = f;
    unsigned r = c.u + 0x7FFFu + ((c.u >> 16) & 1u);   // round-to-nearest-even
    return (u16)(r >> 16);
}

// window token -> original token index (B,H*W layout)
__device__ __forceinline__ int win_tok(int win, int n) {
    int b  = win >> 6;        // 64 windows per batch image
    int wr = win & 63;
    int wh = wr >> 3, ww = wr & 7;
    int ph = n / 7, pw = n % 7;
    return b * 3136 + (wh * 7 + ph) * 56 + (ww * 7 + pw);
}

#if USE_TDM
// ---------------------------------------------------------------------------
// Issue one 2-D TDM tile load: rows x cols (bf16) from row-major global
// (row stride = stride0 elements) into LDS at lds_off, with LDS padding of
// 16B every 64B so the LDS row stride is 40 u16 (bank-conflict free).
// Descriptor layout per CDNA5 ISA sec 8.3/8.4 (D# group0/group1).
// ---------------------------------------------------------------------------
__device__ __forceinline__ void tdm_load_tile(uint32_t lds_off, const u16* gptr,
                                              uint32_t dim0, uint32_t dim1,
                                              uint32_t tile0, uint32_t tile1,
                                              uint32_t stride0) {
    uint64_t ga = (uint64_t)(uintptr_t)gptr;
    v4u g0;
    g0.x = 1u;                                              // count=1, user descriptor
    g0.y = lds_off;                                         // lds_addr (bytes)
    g0.z = (uint32_t)ga;                                    // global_addr[31:0]
    g0.w = ((uint32_t)(ga >> 32) & 0x1FFFFFFu) | (2u << 30);// global_addr[56:32], type=2
    v8i_t g1;
    // data_size=1 (2B), pad_enable, pad_interval=3 (16 DWORDs), pad_amount=3 (4 DWORDs)
    g1[0] = (int)((1u << 16) | (1u << 20) | (3u << 22) | (3u << 25));
    g1[1] = (int)((dim0 & 0xFFFFu) << 16);                  // tensor_dim0[15:0] @ b63:48
    g1[2] = (int)(((dim0 >> 16) & 0xFFFFu) | ((dim1 & 0xFFFFu) << 16)); // dim0 hi, dim1 lo
    g1[3] = (int)(((dim1 >> 16) & 0xFFFFu) | (tile0 << 16));// dim1 hi, tile_dim0 @127:112
    g1[4] = (int)(tile1 & 0xFFFFu);                         // tile_dim1; tile_dim2=0
    g1[5] = (int)stride0;                                   // tensor_dim0_stride[31:0]
    g1[6] = 0;                                              // stride0 hi, stride1 lo
    g1[7] = 0;
    v4i_t g2 = {0, 0, 0, 0};
    v4i_t g3 = {0, 0, 0, 0};
#if defined(__clang_major__) && (__clang_major__ >= 23)
    v8i_t g4 = {0, 0, 0, 0, 0, 0, 0, 0};
    __builtin_amdgcn_tensor_load_to_lds(g0, g1, g2, g3, g4, 0);
#else
    __builtin_amdgcn_tensor_load_to_lds(g0, g1, g2, g3, 0);
#endif
}
#endif

// ---------------------------------------------------------------------------
// fp32 -> bf16 elementwise convert (weights)
// ---------------------------------------------------------------------------
__global__ void cvt_bf16_kernel(const float* __restrict__ in, u16* __restrict__ out, int n) {
    int i = blockIdx.x * blockDim.x + threadIdx.x;
    int stride = gridDim.x * blockDim.x;
    for (; i < n; i += stride) out[i] = f2bf(in[i]);
}

// ---------------------------------------------------------------------------
// LayerNorm over C=384, one wave per token, output bf16
// ---------------------------------------------------------------------------
__global__ void ln_bf16_kernel(const float* __restrict__ x, const float* __restrict__ g,
                               const float* __restrict__ b, u16* __restrict__ out) {
    int lane = threadIdx.x & 31;
    int wave = threadIdx.x >> 5;
    size_t row = (size_t)blockIdx.x * 8 + wave;
    const float* xr = x + row * CDIM;
    float v[12], s = 0.f, s2 = 0.f;
#pragma unroll
    for (int i = 0; i < 12; ++i) { float t = xr[lane + 32 * i]; v[i] = t; s += t; s2 += t * t; }
#pragma unroll
    for (int off = 16; off >= 1; off >>= 1) {
        s  += __shfl_xor(s,  off, 32);
        s2 += __shfl_xor(s2, off, 32);
    }
    float mean = s * (1.f / CDIM);
    float var  = s2 * (1.f / CDIM) - mean * mean;
    float rstd = rsqrtf(var + 1e-5f);
    u16* orow = out + row * CDIM;
#pragma unroll
    for (int i = 0; i < 12; ++i) {
        int c = lane + 32 * i;
        orow[c] = f2bf((v[i] - mean) * rstd * g[c] + b[c]);
    }
}

// ---------------------------------------------------------------------------
// Generic bf16 WMMA GEMM:  out = epilogue(A(MxK) @ W(NxK)^T + bias)
// Tile: 128x128x32, 256 threads (8 waves), each wave -> 2x4 16x16 tiles.
// Tiles staged by TDM (double-buffered) when available.
// MODE 0: bf16 out = acc+bias       (QKV)
// MODE 1: f32  out = resid+acc+bias (proj / fc2 residual)
// MODE 2: bf16 out = gelu(acc+bias) (fc1)
// ---------------------------------------------------------------------------
template <int MODE>
__global__ __launch_bounds__(256)
void gemm_bf16_kernel(const u16* __restrict__ A, const u16* __restrict__ W,
                      const float* __restrict__ bias, const float* __restrict__ resid,
                      void* __restrict__ outp, int M, int N, int K) {
    constexpr int BK  = 32;
    constexpr int LDA = BK + 8;                 // 80B row stride (TDM pad reproduces this)
    constexpr int TILE_U16 = 128 * LDA;
    __shared__ __align__(16) u16 sA[2][TILE_U16];
    __shared__ __align__(16) u16 sB[2][TILE_U16];

    const int tid  = threadIdx.x;
    const int lane = tid & 31;
    const int wave = tid >> 5;
    const int wm   = wave & 3;                  // row group (32 rows)
    const int wn   = wave >> 2;                 // col group (64 cols)
    const int lg   = lane >> 4;
    const int lc   = lane & 15;
    const int m0   = blockIdx.y * 128;
    const int n0   = blockIdx.x * 128;

    v8f acc[2][4];
    const v8f vzero = {0.f, 0.f, 0.f, 0.f, 0.f, 0.f, 0.f, 0.f};
#pragma unroll
    for (int i = 0; i < 2; ++i)
#pragma unroll
        for (int j = 0; j < 4; ++j) acc[i][j] = vzero;

    const int kb = 8 * lg;                      // per-lane K sub-offset for fragments
    const int iters = K / BK;

#if USE_TDM
    const uint32_t saOff = (uint32_t)(uintptr_t)&sA[0][0];
    const uint32_t sbOff = (uint32_t)(uintptr_t)&sB[0][0];
    const uint32_t tileBytes = (uint32_t)(TILE_U16 * 2);
    // prologue: stage tile 0 into buffer 0
    if (wave == 0) {
        tdm_load_tile(saOff, A + (size_t)m0 * K, (uint32_t)K, (uint32_t)M, BK, 128, (uint32_t)K);
        tdm_load_tile(sbOff, W + (size_t)n0 * K, (uint32_t)K, (uint32_t)N, BK, 128, (uint32_t)K);
    }
#endif

    for (int it = 0; it < iters; ++it) {
        const int cur = it & 1;
#if USE_TDM
        if (wave == 0) {
            if (it + 1 < iters) {
                const int nxt = (it + 1) & 1;
                const int kn  = (it + 1) * BK;
                tdm_load_tile(saOff + nxt * tileBytes, A + (size_t)m0 * K + kn,
                              (uint32_t)K, (uint32_t)M, BK, 128, (uint32_t)K);
                tdm_load_tile(sbOff + nxt * tileBytes, W + (size_t)n0 * K + kn,
                              (uint32_t)K, (uint32_t)N, BK, 128, (uint32_t)K);
                __builtin_amdgcn_s_wait_tensorcnt(2);   // tile `it` complete (in-order)
            } else {
                __builtin_amdgcn_s_wait_tensorcnt(0);
            }
        }
        __syncthreads();
#else
        const int k0 = it * BK;
#pragma unroll
        for (int l = 0; l < 2; ++l) {
            int s   = tid + l * 256;
            int row = s >> 2;
            int c8  = (s & 3) * 8;
            const u16* ga = A + (size_t)(m0 + row) * K + k0 + c8;
            const u16* gb = W + (size_t)(n0 + row) * K + k0 + c8;
            *(U4*)&sA[cur][row * LDA + c8] = *(const U4*)ga;
            *(U4*)&sB[cur][row * LDA + c8] = *(const U4*)gb;
            if (k0 + BK < K) {
                __builtin_prefetch(ga + BK, 0, 1);
                __builtin_prefetch(gb + BK, 0, 1);
            }
        }
        __syncthreads();
#endif

        Frag af[2], bfr[4];
#pragma unroll
        for (int i = 0; i < 2; ++i) {
            const u16* p = &sA[cur][(32 * wm + 16 * i + lc) * LDA];
            af[i].q[0] = *(const U4*)(p + kb);
            af[i].q[1] = *(const U4*)(p + kb + 16);
        }
#pragma unroll
        for (int j = 0; j < 4; ++j) {
            const u16* p = &sB[cur][(64 * wn + 16 * j + lc) * LDA];
            bfr[j].q[0] = *(const U4*)(p + kb);
            bfr[j].q[1] = *(const U4*)(p + kb + 16);
        }
#pragma unroll
        for (int i = 0; i < 2; ++i)
#pragma unroll
            for (int j = 0; j < 4; ++j)
                acc[i][j] = __builtin_amdgcn_wmma_f32_16x16x32_bf16(
                    false, af[i].v, false, bfr[j].v, (short)0, acc[i][j], false, false);
        __syncthreads();
    }

    // epilogue — C/D layout: vgpr r, lane: M = r + 8*lg, N = lc (per 16x16 tile)
#pragma unroll
    for (int j = 0; j < 4; ++j) {
        int col = n0 + 64 * wn + 16 * j + lc;
        float bv = bias[col];
#pragma unroll
        for (int i = 0; i < 2; ++i) {
#pragma unroll
            for (int r = 0; r < 8; ++r) {
                size_t row = (size_t)(m0 + 32 * wm + 16 * i + r + 8 * lg);
                float v = acc[i][j][r] + bv;
                if (MODE == 0) {
                    ((u16*)outp)[row * N + col] = f2bf(v);
                } else if (MODE == 1) {
                    ((float*)outp)[row * N + col] = resid[row * N + col] + v;
                } else {
                    float ge = 0.5f * v * (1.f + erff(v * 0.70710678118654752f));
                    ((u16*)outp)[row * N + col] = f2bf(ge);
                }
            }
        }
    }
}

// ---------------------------------------------------------------------------
// Windowed attention, one workgroup per (window, head); 4 waves, each owns a
// 16-row slab of the 64-row padded score matrix (N=49 padded to 64).
// ---------------------------------------------------------------------------
__global__ __launch_bounds__(128)
void attn_kernel(const u16* __restrict__ qkv, const float* __restrict__ rpb,
                 u16* __restrict__ out) {
    const int win  = blockIdx.x;
    const int head = blockIdx.y;
    const int lane = threadIdx.x & 31;
    const int wave = threadIdx.x >> 5;
    const int lg = lane >> 4, lc = lane & 15;
    const int kb = 8 * lg;
    const float scale = 0.17677669529663687f;   // 1/sqrt(32)
    const v8f vzero = {0.f, 0.f, 0.f, 0.f, 0.f, 0.f, 0.f, 0.f};

    __shared__ __align__(16) u16 sP[4][16][72];  // per-wave P slab, padded stride

    // ---- Q fragment (rows 16*wave .. +15, K = d = 32) ----
    Frag qf;
    {
        int nq = 16 * wave + lc;
        if (nq < 49) {
            const u16* p = qkv + (size_t)win_tok(win, nq) * C3 + head * 32;
            qf.q[0] = *(const U4*)(p + kb);
            qf.q[1] = *(const U4*)(p + kb + 16);
        } else {
            U4 z; z.x = z.y = z.z = z.w = 0u; qf.q[0] = z; qf.q[1] = z;
        }
    }
    // ---- K fragments (B operand = rows of K matrix) ----
    Frag kf[4];
#pragma unroll
    for (int ct = 0; ct < 4; ++ct) {
        int nk = 16 * ct + lc;
        if (nk < 49) {
            const u16* p = qkv + (size_t)win_tok(win, nk) * C3 + CDIM + head * 32;
            kf[ct].q[0] = *(const U4*)(p + kb);
            kf[ct].q[1] = *(const U4*)(p + kb + 16);
        } else {
            U4 z; z.x = z.y = z.z = z.w = 0u; kf[ct].q[0] = z; kf[ct].q[1] = z;
        }
    }

    // ---- S = Q K^T : 4 WMMAs (full K=32 per instruction) ----
    v8f s[4];
#pragma unroll
    for (int ct = 0; ct < 4; ++ct)
        s[ct] = __builtin_amdgcn_wmma_f32_16x16x32_bf16(
            false, qf.v, false, kf[ct].v, (short)0, vzero, false, false);

    // ---- scale + relative position bias + mask ----
#pragma unroll
    for (int ct = 0; ct < 4; ++ct) {
        int nk = 16 * ct + lc;
#pragma unroll
        for (int r = 0; r < 8; ++r) {
            int nq = 16 * wave + r + 8 * lg;
            if (nk >= 49) {
                s[ct][r] = -1e30f;
            } else {
                float bv = 0.f;
                if (nq < 49) {
                    int ph0 = nq / 7, pw0 = nq % 7;
                    int ph1 = nk / 7, pw1 = nk % 7;
                    int idx = (ph0 - ph1 + 6) * 13 + (pw0 - pw1 + 6);
                    bv = rpb[idx * HEADS + head];
                }
                s[ct][r] = s[ct][r] * scale + bv;
            }
        }
    }

    // ---- row softmax over 64 cols (4 tiles x 16 lanes) ----
#pragma unroll
    for (int r = 0; r < 8; ++r) {
        float m = fmaxf(fmaxf(s[0][r], s[1][r]), fmaxf(s[2][r], s[3][r]));
#pragma unroll
        for (int off = 8; off >= 1; off >>= 1) m = fmaxf(m, __shfl_xor(m, off, 16));
        float sum = 0.f;
#pragma unroll
        for (int ct = 0; ct < 4; ++ct) { float e = __expf(s[ct][r] - m); s[ct][r] = e; sum += e; }
#pragma unroll
        for (int off = 8; off >= 1; off >>= 1) sum += __shfl_xor(sum, off, 16);
        float inv = 1.f / sum;
#pragma unroll
        for (int ct = 0; ct < 4; ++ct) s[ct][r] *= inv;
    }

    // ---- re-swizzle P (C layout -> A layout) through LDS ----
#pragma unroll
    for (int ct = 0; ct < 4; ++ct)
#pragma unroll
        for (int r = 0; r < 8; ++r)
            sP[wave][r + 8 * lg][16 * ct + lc] = f2bf(s[ct][r]);

    // ---- O = P V : 2 K-chunks x 2 col tiles = 4 WMMAs ----
    v8f o[2] = {vzero, vzero};
#pragma unroll
    for (int kc = 0; kc < 2; ++kc) {
        Frag pf;
        pf.q[0] = *(const U4*)&sP[wave][lc][32 * kc + kb];
        pf.q[1] = *(const U4*)&sP[wave][lc][32 * kc + 16 + kb];
#pragma unroll
        for (int cto = 0; cto < 2; ++cto) {
            Frag vf;
            int dcol = head * 32 + 16 * cto + lc;
#pragma unroll
            for (int j = 0; j < 8; ++j) {
                int k = 32 * kc + kb + j;
                vf.u[j] = (k < 49) ? qkv[(size_t)win_tok(win, k) * C3 + 2 * CDIM + dcol] : (u16)0;
            }
#pragma unroll
            for (int j = 0; j < 8; ++j) {
                int k = 32 * kc + 16 + kb + j;
                vf.u[8 + j] = (k < 49) ? qkv[(size_t)win_tok(win, k) * C3 + 2 * CDIM + dcol] : (u16)0;
            }
            o[cto] = __builtin_amdgcn_wmma_f32_16x16x32_bf16(
                false, pf.v, false, vf.v, (short)0, o[cto], false, false);
        }
    }

    // ---- scatter output rows back to original token order (bf16) ----
#pragma unroll
    for (int cto = 0; cto < 2; ++cto) {
#pragma unroll
        for (int r = 0; r < 8; ++r) {
            int nq = 16 * wave + r + 8 * lg;
            if (nq < 49)
                out[(size_t)win_tok(win, nq) * CDIM + head * 32 + 16 * cto + lc] = f2bf(o[cto][r]);
        }
    }
}

// ---------------------------------------------------------------------------
// Launcher
// ---------------------------------------------------------------------------
extern "C" void kernel_launch(void* const* d_in, const int* in_sizes, int n_in,
                              void* d_out, int out_size, void* d_ws, size_t ws_size,
                              hipStream_t stream) {
    (void)in_sizes; (void)n_in; (void)out_size; (void)ws_size;

    const float* x     = (const float*)d_in[0];
    const float* n1g   = (const float*)d_in[1];
    const float* n1b   = (const float*)d_in[2];
    const float* qkvw  = (const float*)d_in[3];
    const float* qkvb  = (const float*)d_in[4];
    const float* rpb   = (const float*)d_in[5];
    const float* projw = (const float*)d_in[6];
    const float* projb = (const float*)d_in[7];
    const float* n2g   = (const float*)d_in[8];
    const float* n2b   = (const float*)d_in[9];
    const float* fc1w  = (const float*)d_in[10];
    const float* fc1b  = (const float*)d_in[11];
    const float* fc2w  = (const float*)d_in[12];
    const float* fc2b  = (const float*)d_in[13];

    size_t off = 0;
    auto take = [&](size_t bytes) -> char* {
        char* p = (char*)d_ws + off;
        off += (bytes + 255) & ~(size_t)255;
        return p;
    };
    u16*   xn  = (u16*)  take((size_t)MTOK * CDIM * 2);
    u16*   wq  = (u16*)  take((size_t)C3   * CDIM * 2);
    u16*   wp  = (u16*)  take((size_t)CDIM * CDIM * 2);
    u16*   w1  = (u16*)  take((size_t)HID  * CDIM * 2);
    u16*   w2  = (u16*)  take((size_t)CDIM * HID  * 2);
    u16*   qkv = (u16*)  take((size_t)MTOK * C3   * 2);
    u16*   ao  = (u16*)  take((size_t)MTOK * CDIM * 2);
    float* x2  = (float*)take((size_t)MTOK * CDIM * 4);
    u16*   y2  = (u16*)  take((size_t)MTOK * CDIM * 2);
    u16*   h1  = (u16*)  take((size_t)MTOK * HID  * 2);

    // weights fp32 -> bf16
    cvt_bf16_kernel<<<512, 256, 0, stream>>>(qkvw,  wq, C3 * CDIM);
    cvt_bf16_kernel<<<512, 256, 0, stream>>>(projw, wp, CDIM * CDIM);
    cvt_bf16_kernel<<<512, 256, 0, stream>>>(fc1w,  w1, HID * CDIM);
    cvt_bf16_kernel<<<512, 256, 0, stream>>>(fc2w,  w2, CDIM * HID);

    // LN1 -> bf16
    ln_bf16_kernel<<<MTOK / 8, 256, 0, stream>>>(x, n1g, n1b, xn);

    // QKV GEMM (bias -> bf16)
    gemm_bf16_kernel<0><<<dim3(C3 / 128, MTOK / 128), 256, 0, stream>>>(
        xn, wq, qkvb, nullptr, qkv, MTOK, C3, CDIM);

    // windowed attention
    attn_kernel<<<dim3(NWIN, HEADS), 128, 0, stream>>>(qkv, rpb, ao);

    // proj GEMM + residual (x2 = x + attn_proj)
    gemm_bf16_kernel<1><<<dim3(CDIM / 128, MTOK / 128), 256, 0, stream>>>(
        ao, wp, projb, x, x2, MTOK, CDIM, CDIM);

    // LN2 -> bf16
    ln_bf16_kernel<<<MTOK / 8, 256, 0, stream>>>(x2, n2g, n2b, y2);

    // fc1 GEMM + exact GELU -> bf16
    gemm_bf16_kernel<2><<<dim3(HID / 128, MTOK / 128), 256, 0, stream>>>(
        y2, w1, fc1b, nullptr, h1, MTOK, HID, CDIM);

    // fc2 GEMM + residual -> d_out (fp32)
    gemm_bf16_kernel<1><<<dim3(CDIM / 128, MTOK / 128), 256, 0, stream>>>(
        h1, w2, fc2b, x2, (float*)d_out, MTOK, CDIM, HID);
}
